// QuantizedComplexConv2d_72808285601855
// MI455X (gfx1250) — matrix operand order repaired and checked by reading, not compiled, not run on hardware
//
#include <hip/hip_runtime.h>
#include <hip/hip_bf16.h>

// ---------------------------------------------------------------------------
// QuantizedComplexConv2d for MI455X (gfx1250): implicit GEMM on
// v_wmma_f32_16x16x32_bf16.
//
//   Combined GEMM:  D[128 x Npix] = Wc[128 x 576] * Im2col[576 x Npix] + bias
//     rows 0..63   -> out_real  = [ wr | -wi ] . [xr ; xi]
//     rows 64..127 -> out_imag  = [ wi |  wr ] . [xr ; xi]
//   K ordering: k = (kh*3+kw)*64 + c'   (c' = 0..31 real ch, 32..63 imag ch)
// ---------------------------------------------------------------------------

typedef __attribute__((ext_vector_type(16))) __bf16 bhalf16;
typedef __attribute__((ext_vector_type(8)))  __bf16 bhalf8;
typedef __attribute__((ext_vector_type(8)))  float  v8f;

union BF16x16 { bhalf16 v; struct { bhalf8 lo; bhalf8 hi; } p; };

__device__ __forceinline__ __bf16 f2bf(float x) {
    unsigned int u = __builtin_bit_cast(unsigned int, x);
    u += 0x7FFFu + ((u >> 16) & 1u);              // round to nearest even
    unsigned short s = (unsigned short)(u >> 16);
    return __builtin_bit_cast(__bf16, s);
}

#define CIN   32
#define COUT  64
#define HW    256
#define KPOS  9
#define MROWS 128                 // 2*COUT (real rows then imag rows)
#define KTOT  576                 // 9 taps * 64 combined channels
#define TH    8                   // output rows per block
#define TW    32                  // output cols per block
#define IR    (TH + 2)            // 10
#define IC    (TW + 2)            // 34
#define CH    64                  // combined input channels (xr:0..31, xi:32..63)
#define OUT_IMAG_OFF (4 * COUT * HW * HW)   // 16,777,216 floats

// ---------------------------------------------------------------------------
// Prep: build combined bf16 weight matrix Wc[128][576] in workspace.
// qw_* layout: [O=64][C=32][3][3] fp32, flat ((m*32+c)*9 + pos).
// ---------------------------------------------------------------------------
__global__ void qcc_prep_weights(const float* __restrict__ wr,
                                 const float* __restrict__ wi,
                                 __bf16* __restrict__ Wc) {
    int idx = blockIdx.x * 256 + threadIdx.x;
    if (idx >= MROWS * KTOT) return;
    int m   = idx / KTOT;
    int k   = idx % KTOT;
    int pos = k / CH;
    int cp  = k % CH;
    int c   = cp & 31;
    float v;
    if (m < COUT) {                       // real output rows: [wr | -wi]
        int o = (m * CIN + c) * KPOS + pos;
        v = (cp < CIN) ? wr[o] : -wi[o];
    } else {                              // imag output rows: [wi |  wr]
        int o = ((m - COUT) * CIN + c) * KPOS + pos;
        v = (cp < CIN) ? wi[o] : wr[o];
    }
    Wc[idx] = f2bf(v);
}

// ---------------------------------------------------------------------------
// Main implicit-GEMM kernel. Block = 256 threads = 8 waves.
// Block tile: one batch, TH x TW = 8x32 output pixels, all 128 M rows.
// Wave w owns M rows [16w, 16w+16); loops over 16 pixel groups of 16.
// ---------------------------------------------------------------------------
__global__ void __launch_bounds__(256)
qcc_conv_wmma(const float* __restrict__ xr,
              const float* __restrict__ xi,
              const __bf16* __restrict__ Wc,
              const float* __restrict__ bias_r,
              const float* __restrict__ bias_i,
              float* __restrict__ out) {
    __shared__ __attribute__((aligned(32))) __bf16 smem[IR * IC * CH]; // 43.5 KB

    const int n  = blockIdx.z;
    const int h0 = blockIdx.y * TH;
    const int w0 = blockIdx.x * TW;

    const int tid  = threadIdx.x;
    const int wave = tid >> 5;
    const int lane = tid & 31;

    // ---- stage input tile (bf16, zero-padded halo) into LDS ----
    // layout: smem[(r*IC + cc)*CH + c], c innermost -> lane-contiguous writes
    const int total = IR * IC * CH;          // 21760 = 85 * 256
    for (int e = tid; e < total; e += 256) {
        int c  = e & (CH - 1);
        int rc = e >> 6;
        int r  = rc / IC;
        int cc = rc - r * IC;
        int h  = h0 + r - 1;
        int w  = w0 + cc - 1;
        float v = 0.0f;
        if ((unsigned)h < HW && (unsigned)w < HW) {
            const float* src = (c < CIN) ? xr : xi;
            v = src[((n * CIN + (c & 31)) * HW + h) * HW + w];
        }
        smem[e] = f2bf(v);
    }

    // ---- preload all 18 A-fragments (weights) into registers ----
    // A frag, contiguous k order (B uses same): lane<16 -> K k0+0..15, lane>=16 -> K k0+16..31
    const int m0    = wave * 16;
    const int arow  = m0 + (lane & 15);
    const int khalf = (lane >> 4) << 4;      // 0 or 16
    bhalf16 afrag[18];
#pragma unroll
    for (int ks = 0; ks < 18; ks++) {
        const __bf16* pa = Wc + arow * KTOT + ks * 32 + khalf;
        afrag[ks] = *(const bhalf16*)(pa);    // K k0+khalf+0..15
    }

    // ---- bias into initial accumulator: acc VGPR j -> row m0+j(+8 hi lanes)
    const int mhi = (lane >> 4) << 3;        // 0 or 8
    v8f cinit;
#pragma unroll
    for (int j = 0; j < 8; j++) {
        int m = m0 + j + mhi;
        cinit[j] = (m < COUT) ? bias_r[m] : bias_i[m - COUT];
    }

    __syncthreads();

    // ---- 16 pixel groups of 16 columns each (2 groups per output row) ----
    const int bhalf = (lane >> 4) << 4;      // B frag: 16 consecutive channels
    for (int g = 0; g < 16; g++) {
        const int orow = g >> 1;
        const int ocol = ((g & 1) << 4) + (lane & 15);
        v8f acc = cinit;
#pragma unroll
        for (int pos = 0; pos < KPOS; pos++) {
            const int kh = pos / 3;
            const int kw = pos - kh * 3;
            const int r  = orow + kh;
            const int cc = ocol + kw;
            const __bf16* bp = smem + (r * IC + cc) * CH + bhalf;
            bhalf16 b0 = *(const bhalf16*)(bp);        // channels 0..31 half
            bhalf16 b1 = *(const bhalf16*)(bp + 32);   // channels 32..63 half
            acc = __builtin_amdgcn_wmma_f32_16x16x32_bf16(
                      false, afrag[2 * pos],     false, b0, (short)0, acc, false, false);
            acc = __builtin_amdgcn_wmma_f32_16x16x32_bf16(
                      false, afrag[2 * pos + 1], false, b1, (short)0, acc, false, false);
        }
        // ---- store 16x16 fp32 tile (C/D layout: VGPR j -> M=j / j+8) ----
        const int h = h0 + orow;
        const int w = w0 + ((g & 1) << 4) + (lane & 15);
#pragma unroll
        for (int j = 0; j < 8; j++) {
            int m = m0 + j + mhi;
            float* op = (m < COUT)
                ? out + (((n * COUT + m) * HW + h) * HW + w)
                : out + OUT_IMAG_OFF + (((n * COUT + (m - COUT)) * HW + h) * HW + w);
            *op = acc[j];
        }
    }
}

// ---------------------------------------------------------------------------
// Launch: inputs in dict order:
//   0 input_real [4,32,256,256] f32    1 input_imag [4,32,256,256] f32
//   2 qw_real    [64,32,3,3]    f32    3 qw_imag    [64,32,3,3]    f32
//   4 bias_real  [64]           f32    5 bias_imag  [64]           f32
// d_out: out_real then out_imag, each [4,64,256,256] f32.
// ---------------------------------------------------------------------------
extern "C" void kernel_launch(void* const* d_in, const int* in_sizes, int n_in,
                              void* d_out, int out_size, void* d_ws, size_t ws_size,
                              hipStream_t stream) {
    const float* xr = (const float*)d_in[0];
    const float* xi = (const float*)d_in[1];
    const float* wr = (const float*)d_in[2];
    const float* wi = (const float*)d_in[3];
    const float* br = (const float*)d_in[4];
    const float* bi = (const float*)d_in[5];
    float* out = (float*)d_out;
    __bf16* Wc = (__bf16*)d_ws;   // 128*576*2 = 147,456 bytes

    const int wtot = MROWS * KTOT;
    qcc_prep_weights<<<(wtot + 255) / 256, 256, 0, stream>>>(wr, wi, Wc);

    dim3 grid(HW / TW, HW / TH, 4);   // (8, 32, 4)
    qcc_conv_wmma<<<grid, 256, 0, stream>>>(xr, xi, Wc, br, bi, out);
}
